// VQVAE_70248485094037
// MI455X (gfx1250) — matrix operand order repaired and checked by reading, not compiled, not run on hardware
//
#include <hip/hip_runtime.h>
#include <hip/hip_bf16.h>

typedef __attribute__((ext_vector_type(16))) __bf16 bf16x16;
typedef __attribute__((ext_vector_type(8)))  float  floatx8;

#define BATCH 32

__device__ __forceinline__ unsigned short f2bf(float f) {
  union { float f; unsigned u; } v; v.f = f;
  unsigned r = v.u + 0x7FFFu + ((v.u >> 16) & 1u);   // round-to-nearest-even
  return (unsigned short)(r >> 16);
}
__device__ __forceinline__ float bf2f(unsigned short h) {
  union { unsigned u; float f; } v; v.u = ((unsigned)h) << 16; return v.f;
}

// ---------------------------------------------------------------------------
// Implicit-GEMM convolution / transposed convolution.
//   M = 32 output channels / block (2 WMMA tiles), N = 128 pixels / block
//   (16 / wave), K = Cin*KH*KW consumed 32 at a time via
//   v_wmma_f32_16x16x32_bf16 with fp32 accumulation. Two WMMAs share each
//   staged B tile, halving gather/convert/LDS cost per FLOP.
// ---------------------------------------------------------------------------
template <bool TR, bool INF32, bool OUTF32>
__global__ __launch_bounds__(256) void conv_wmma(
    const void* __restrict__ in,
    const float* __restrict__ w, const float* __restrict__ bias,
    const unsigned short* __restrict__ res,            // bf16 residual or null
    void* __restrict__ out,
    int Cin, int Hin, int Win, int OC, int OH, int OW,
    int KH, int KW, int stride, int pad,
    int relu_in, int relu_out, int tanh_out)
{
  __shared__ __align__(16) unsigned short Alds[32 * 34];   // weights [m][k], pitch 34
  __shared__ __align__(16) unsigned short Blds[128 * 34];  // inputs  [n][k], pitch 34

  const int tid  = threadIdx.x;
  const int lane = tid & 31;
  const int wave = tid >> 5;
  const int half = lane >> 4;     // 0: lanes 0-15, 1: lanes 16-31
  const int mrow = lane & 15;

  const int Ntot    = BATCH * OH * OW;
  const int n0      = blockIdx.x * 128;
  const int oc0     = blockIdx.y * 32;
  const int HWin    = Hin * Win;
  const int nTaps   = KH * KW;
  const int nChunks = (Cin + 31) >> 5;

  // ---- per-thread staging geometry (loop-invariant) ----
  // B staging: this thread always stages pixel p = tid&127, channels kkB+2j.
  const int  pB  = tid & 127;
  const int  kkB = tid >> 7;                 // 0 or 1
  const int  nB  = n0 + pB;
  const bool nBok = (nB < Ntot);
  int bB, ohB, owB;
  {
    const int n = nBok ? nB : 0;
    owB = n % OW;
    const int t = n / OW;
    ohB = t % OH;
    bB  = t / OH;
  }
  unsigned short* BldsRow = &Blds[pB * 34 + kkB];

  // A staging: thread stages (mA + 8*t2, kkA), t2 = 0..3.
  const int mA  = tid >> 5;                  // 0..7
  const int kkA = tid & 31;

  // fragment read bases (constant offsets fold into ds_load immediates)
  const unsigned short* aRow0 = &Alds[mrow * 34 + (half ? 8 : 0)];
  const unsigned short* aRow1 = aRow0 + 16 * 34;
  const unsigned short* bRow  = &Blds[(wave * 16 + mrow) * 34 + (half ? 8 : 0)];

  floatx8 acc0 = {0.f, 0.f, 0.f, 0.f, 0.f, 0.f, 0.f, 0.f};
  floatx8 acc1 = {0.f, 0.f, 0.f, 0.f, 0.f, 0.f, 0.f, 0.f};

  for (int kh = 0; kh < KH; ++kh) {
    for (int kw = 0; kw < KW; ++kw) {
      // per-tap geometry for this thread's B pixel
      int ih, iw; bool tapok;
      if (!TR) {
        ih = ohB * stride - pad + kh;
        iw = owB * stride - pad + kw;
        tapok = true;
      } else {                                // gather form of ConvTranspose2d
        const int th = ohB + pad - kh;
        const int tw = owB + pad - kw;
        tapok = (th >= 0) && (tw >= 0) &&
                (th % stride == 0) && (tw % stride == 0);
        ih = th / stride; iw = tw / stride;
      }
      const bool bvalid = nBok && tapok &&
                          ih >= 0 && ih < Hin && iw >= 0 && iw < Win;
      const long long ibase =
          (long long)bB * Cin * HWin + (long long)ih * Win + iw;
      const int wtap = kh * KW + kw;

      for (int chunk = 0; chunk < nChunks; ++chunk) {
        const int ic0 = chunk << 5;
        __syncthreads();   // previous iteration's fragment reads done

        // ---- stage A: 4 weight elements / thread (32 oc x 32 ic tile) ----
        {
          const int  ic   = ic0 + kkA;
          const bool icok = (ic < Cin);
#pragma unroll
          for (int t2 = 0; t2 < 4; ++t2) {
            const int m  = mA + t2 * 8;
            const int oc = oc0 + m;
            float v = 0.f;
            if (icok && oc < OC) {
              const long long widx = TR
                ? (((long long)ic * OC  + oc) * nTaps + wtap)   // (in,out,kh,kw)
                : (((long long)oc * Cin + ic) * nTaps + wtap);  // (out,in,kh,kw)
              v = w[widx];
            }
            Alds[m * 34 + kkA] = f2bf(v);
          }
        }
        // ---- stage B: 16 channels / thread, incremental addressing ----
        {
          const bool fullChunk = (ic0 + 32 <= Cin);
          long long addr = ibase + (long long)(ic0 + kkB) * HWin;
#pragma unroll
          for (int j = 0; j < 16; ++j) {
            float v = 0.f;
            if (bvalid && (fullChunk || (ic0 + kkB + 2 * j) < Cin)) {
              v = INF32 ? ((const float*)in)[addr]
                        : bf2f(((const unsigned short*)in)[addr]);
              if (relu_in) v = fmaxf(v, 0.f);
            }
            BldsRow[2 * j] = f2bf(v);
            addr += 2ll * HWin;
          }
        }
        __syncthreads();

        // ---- fragments per CDNA5 16-bit A(16x32)/B(32x16) layouts ----
        union { unsigned u[8]; bf16x16 v; } af0, af1, bfr;
#pragma unroll
        for (int j = 0; j < 8; ++j) {
          const int ka = (j < 4) ? (2 * j) : (16 + 2 * (j - 4));
          af0.u[j] = *reinterpret_cast<const unsigned*>(aRow0 + ka);
          af1.u[j] = *reinterpret_cast<const unsigned*>(aRow1 + ka);
          bfr.u[j] = *reinterpret_cast<const unsigned*>(bRow + ka);
        }
        acc0 = __builtin_amdgcn_wmma_f32_16x16x32_bf16(false, af0.v, false, bfr.v,
                                                       (short)0, acc0, false, false);
        acc1 = __builtin_amdgcn_wmma_f32_16x16x32_bf16(false, af1.v, false, bfr.v,
                                                       (short)0, acc1, false, false);
      }
    }
  }

  // ---- epilogue: C layout vgpr r -> M = r + 8*half, N = lane&15 ----
  const int n = n0 + wave * 16 + mrow;
  if (n < Ntot) {
    const int ow = n % OW;
    const int t1 = n / OW;
    const int oh = t1 % OH;
    const int b  = t1 / OH;
    const long long base =
        (long long)b * OC * (long long)(OH * OW) + (long long)oh * OW + ow;
#pragma unroll
    for (int r = 0; r < 8; ++r) {
      const int m = r + (half ? 8 : 0);
#pragma unroll
      for (int t2 = 0; t2 < 2; ++t2) {
        const int oc = oc0 + m + t2 * 16;
        if (oc < OC) {
          float v = (t2 ? acc1[r] : acc0[r]) + bias[oc];
          const long long oidx = base + (long long)oc * (OH * OW);
          if (res)      v += bf2f(res[oidx]);
          if (relu_out) v = fmaxf(v, 0.f);
          if (tanh_out) v = tanhf(v);
          if (OUTF32) ((float*)out)[oidx] = v;
          else        ((unsigned short*)out)[oidx] = f2bf(v);
        }
      }
    }
  }
}

// ---------------------------------------------------------------------------
// VQ: one wave per pixel; lanes stride the 512 codes; butterfly min-reduce.
// Writes argmin index and straight-through z_q (bf16, NCHW).
// ---------------------------------------------------------------------------
__global__ __launch_bounds__(256) void vq_argmin(
    const float* __restrict__ ze, const float* __restrict__ cb,
    int* __restrict__ idx, unsigned short* __restrict__ zq,
    int Npix, int HW)
{
  __shared__ float zl[8][64];
  const int lane = threadIdx.x & 31;
  const int wave = threadIdx.x >> 5;
  int p = blockIdx.x * 8 + wave;
  const bool live = (p < Npix);
  if (!live) p = Npix - 1;
  const int b = p / HW;
  const int s = p - b * HW;

  for (int d = lane; d < 64; d += 32)
    zl[wave][d] = ze[((long long)b * 64 + d) * HW + s];
  __syncthreads();

  float best = 3.4e38f; int bi = 0;
  for (int c = lane; c < 512; c += 32) {
    const float* cbc = cb + c * 64;
    float dsum = 0.f;
#pragma unroll 8
    for (int d = 0; d < 64; ++d) {
      const float df = zl[wave][d] - cbc[d];
      dsum += df * df;
    }
    if (dsum < best) { best = dsum; bi = c; }
  }
  for (int off = 16; off; off >>= 1) {       // wave32 butterfly
    const float ob = __shfl_xor(best, off, 32);
    const int   oi = __shfl_xor(bi,   off, 32);
    if (ob < best || (ob == best && oi < bi)) { best = ob; bi = oi; }
  }
  if (live) {
    if (lane == 0) idx[p] = bi;
    for (int d = lane; d < 64; d += 32)
      zq[((long long)b * 64 + d) * HW + s] = f2bf(cb[bi * 64 + d]);
  }
}

// loss = (1+BETA) * mean((z_e - cb[idx])^2)   (cb & commit terms identical fwd)
__global__ __launch_bounds__(256) void vq_loss(
    const float* __restrict__ ze, const float* __restrict__ cb,
    const int* __restrict__ idx, float* __restrict__ loss,
    int Npix, int HW, float scale)
{
  const int tot = Npix * 64;
  float sum = 0.f;
  for (int e = blockIdx.x * blockDim.x + threadIdx.x; e < tot;
       e += gridDim.x * blockDim.x) {
    const int p = e >> 6, d = e & 63;
    const int b = p / HW, s = p - b * HW;
    const float zev = ze[((long long)b * 64 + d) * HW + s];
    const float zqv = cb[idx[p] * 64 + d];
    const float df  = zev - zqv;
    sum += df * df;
  }
  __shared__ float red[256];
  red[threadIdx.x] = sum;
  __syncthreads();
  for (int s2 = 128; s2; s2 >>= 1) {
    if (threadIdx.x < (unsigned)s2) red[threadIdx.x] += red[threadIdx.x + s2];
    __syncthreads();
  }
  if (threadIdx.x == 0) atomicAdd(loss, red[0] * scale);
}

// ---------------------------------------------------------------------------
extern "C" void kernel_launch(void* const* d_in, const int* in_sizes, int n_in,
                              void* d_out, int out_size, void* d_ws, size_t ws_size,
                              hipStream_t stream) {
  (void)in_sizes; (void)n_in; (void)out_size; (void)ws_size;

  auto W  = [&](int i) { return (const float*)d_in[i]; };
  const float* x  = W(0);
  const float* CB = W(23);

  // workspace layout (bf16 activations; bytes)
  char* ws = (char*)d_ws;
  unsigned short* R0 = (unsigned short*)(ws);                 // 32*256*68*68*2 = 75,759,616
  unsigned short* R1 = (unsigned short*)(ws +  75759616);     // 32*256*34*34*2 = 18,939,904
  unsigned short* R2 = (unsigned short*)(ws +  94699520);     // 18,939,904
  float*          ZE = (float*)        (ws + 113639424);      // 32*64*34*34*4 = 9,469,952
  int*           IDX = (int*)          (ws + 123109376);      // 36992*4
  unsigned short* ZQ = (unsigned short*)(ws + 123257344);     // 32*64*34*34*2

  float* OUT  = (float*)d_out;
  float* LOSS = OUT + 32 * 3 * 136 * 136;

  auto conv = [&](const void* in, int inf, const float* w, const float* b,
                  const unsigned short* res, void* out, int outf,
                  int Cin, int Hin, int Win, int OC, int OH, int OW,
                  int KH, int KW, int st, int pd, int tr, int ri, int ro, int to) {
    const int Ntot = BATCH * OH * OW;
    dim3 g((Ntot + 127) / 128, (OC + 31) / 32);
#define LAUNCH_CONV(T_, I_, O_)                                            \
    conv_wmma<T_, I_, O_><<<g, 256, 0, stream>>>(in, w, b, res, out,       \
        Cin, Hin, Win, OC, OH, OW, KH, KW, st, pd, ri, ro, to)
    if (tr) { if (inf) { if (outf) LAUNCH_CONV(true,  true,  true);
                         else      LAUNCH_CONV(true,  true,  false); }
              else     { if (outf) LAUNCH_CONV(true,  false, true);
                         else      LAUNCH_CONV(true,  false, false); } }
    else    { if (inf) { if (outf) LAUNCH_CONV(false, true,  true);
                         else      LAUNCH_CONV(false, true,  false); }
              else     { if (outf) LAUNCH_CONV(false, false, true);
                         else      LAUNCH_CONV(false, false, false); } }
#undef LAUNCH_CONV
  };

  // ---- encoder ----
  conv(x,  1, W(1),  W(2),  nullptr, R0, 0,   3, 128, 128, 256, 64, 64, 4, 4, 2, 1, 0, 0, 1, 0); // relu(conv s2)
  conv(R0, 0, W(3),  W(4),  nullptr, R1, 0, 256,  64,  64, 256, 32, 32, 4, 4, 2, 1, 0, 0, 0, 0); // conv s2
  conv(R1, 0, W(5),  W(6),  nullptr, R2, 0, 256,  32,  32, 256, 32, 32, 3, 3, 1, 1, 0, 1, 0, 0); // res1 3x3(relu)
  conv(R2, 0, W(7),  W(8),  R1,      R0, 0, 256,  32,  32, 256, 32, 32, 1, 1, 1, 0, 0, 1, 0, 0); // res1 1x1 + skip
  conv(R0, 0, W(9),  W(10), nullptr, R1, 0, 256,  32,  32, 256, 32, 32, 3, 3, 1, 1, 0, 1, 0, 0); // res2 3x3(relu)
  conv(R1, 0, W(11), W(12), R0,      R2, 0, 256,  32,  32, 256, 32, 32, 1, 1, 1, 0, 0, 1, 0, 0); // res2 1x1 + skip
  conv(R2, 0, W(21), W(22), nullptr, ZE, 1, 256,  32,  32,  64, 34, 34, 1, 1, 1, 1, 0, 0, 0, 0); // to_lat 1x1 pad1

  // ---- VQ ----
  vq_argmin<<<36992 / 8, 256, 0, stream>>>(ZE, CB, IDX, ZQ, 36992, 1156);

  // ---- decoder ----
  conv(ZQ, 0, W(24), W(25), nullptr, R0, 0,  64, 34, 34, 256,  34,  34, 3, 3, 1, 1, 1, 0, 0, 0); // convT 3x3
  conv(R0, 0, W(13), W(14), nullptr, R1, 0, 256, 34, 34, 256,  34,  34, 3, 3, 1, 1, 0, 1, 0, 0); // res1 3x3(relu)
  conv(R1, 0, W(15), W(16), R0,      R2, 0, 256, 34, 34, 256,  34,  34, 1, 1, 1, 0, 0, 1, 0, 0); // res1 1x1 + skip
  conv(R2, 0, W(17), W(18), nullptr, R0, 0, 256, 34, 34, 256,  34,  34, 3, 3, 1, 1, 0, 1, 0, 0); // res2 3x3(relu)
  conv(R0, 0, W(19), W(20), R2,      R1, 0, 256, 34, 34, 256,  34,  34, 1, 1, 1, 0, 0, 1, 0, 0); // res2 1x1 + skip
  conv(R1, 0, W(26), W(27), nullptr, R0, 0, 256, 34, 34, 256,  68,  68, 4, 4, 2, 1, 1, 1, 1, 0); // relu(convT(relu))
  conv(R0, 0, W(28), W(29), nullptr, OUT,1, 256, 68, 68,   3, 136, 136, 4, 4, 2, 1, 1, 0, 0, 1); // tanh(convT)

  // ---- VQ loss ----
  hipMemsetAsync(LOSS, 0, sizeof(float), stream);
  const float scale = 1.25f / (36992.0f * 64.0f);   // (1+BETA)/Nel
  vq_loss<<<1024, 256, 0, stream>>>(ZE, CB, IDX, LOSS, 36992, 1156, scale);
}
